// GNNSceneEmbeddingNetwork_LearnedEdgeVector_73701638800220
// MI455X (gfx1250) — matrix operand order repaired and checked by reading, not compiled, not run on hardware
//
#include <hip/hip_runtime.h>
#include <hip/hip_bf16.h>

typedef __attribute__((ext_vector_type(16))) _Float16 v16h;
typedef __attribute__((ext_vector_type(8)))  _Float16 v8h;
typedef __attribute__((ext_vector_type(8)))  float    v8f;

#define NODE_DIM 128
#define NEG_SLOPE 0.2f

__device__ __forceinline__ float leaky(float a) { return a > 0.0f ? a : NEG_SLOPE * a; }

// monotonic float<->uint mapping for atomicMax on signed floats
__device__ __forceinline__ unsigned f2ord(float f) {
  unsigned u = __float_as_uint(f);
  return (u & 0x80000000u) ? ~u : (u | 0x80000000u);
}
__device__ __forceinline__ float ord2f(unsigned u) {
  u = (u & 0x80000000u) ? (u & 0x7fffffffu) : ~u;
  return __uint_as_float(u);
}

#define AF(a0, a1) __builtin_shufflevector(a0, a1, 0, 1, 2, 3, 4, 5, 6, 7, \
                                           8, 9, 10, 11, 12, 13, 14, 15)
#define WMMA(af, bf, acc) __builtin_amdgcn_wmma_f32_16x16x32_f16( \
    false, af, false, bf, (short)0, acc, false, false)

// ---------------------------------------------------------------------------
// WMMA GEMM on pre-converted f16 operands, zero padding pre-applied.
//   Ah: [Mp, Kp] f16 row-major (Kp % 32 == 0)
//   Wt: [Np, Kp] f16 = W^T     (Np % 128 == 0, zero-padded rows)
//   C (f32) and/or Ch (f16): [M, N] row-major outputs
// Block = 128 threads = 4 waves; wave owns 32 cols (2 accumulators, A reused);
// block tile = 16 rows x 128 cols. Software-pipelined K loop with running
// pointers (64 B/step) so loads use immediate offsets, not recomputed addrs.
// ---------------------------------------------------------------------------
__global__ __launch_bounds__(128) void k_gemm_wmma(
    const _Float16* __restrict__ Ah, const _Float16* __restrict__ Wt,
    const float* __restrict__ bias, float* __restrict__ C,
    _Float16* __restrict__ Ch, int M, int N, int Kp, int act_relu) {
  const int tid = threadIdx.x;
  const int wave = tid >> 5;
  const int lane = tid & 31;
  const int m0 = blockIdx.x * 16;
  const int n0 = blockIdx.y * 128 + wave * 32;
  const int r = lane & 15;
  const int g = lane >> 4;

  // running element pointers; one K-step (32 halfs) = 64 bytes per stream
  const _Float16* pa  = Ah + (size_t)(m0 + r) * Kp + 16 * g;
  const _Float16* pb0 = Wt + (size_t)(n0 + r) * Kp + 16 * g;
  const _Float16* pb1 = Wt + (size_t)(n0 + 16 + r) * Kp + 16 * g;

  v8f acc0 = {}, acc1 = {};
  // prologue: preload K-step 0
  v8h  a0 = *(const v8h*)pa;
  v8h  a1 = *(const v8h*)(pa + 8);
  v16h b0 = *(const v16h*)pb0;
  v16h b1 = *(const v16h*)pb1;
  pa += 32; pb0 += 32; pb1 += 32;
#pragma unroll 2
  for (int kb = 32; kb < Kp; kb += 32) {
    v8h  na0 = *(const v8h*)pa;
    v8h  na1 = *(const v8h*)(pa + 8);
    v16h nb0 = *(const v16h*)pb0;
    v16h nb1 = *(const v16h*)pb1;
    pa += 32; pb0 += 32; pb1 += 32;
    v16h af = AF(a0, a1);
    acc0 = WMMA(af, b0, acc0);
    acc1 = WMMA(af, b1, acc1);
    a0 = na0; a1 = na1; b0 = nb0; b1 = nb1;
  }
  {
    v16h af = AF(a0, a1);
    acc0 = WMMA(af, b0, acc0);
    acc1 = WMMA(af, b1, acc1);
  }
  // store: VGPR v -> row m0 + v + 8g, cols n0 + r and n0 + 16 + r
#pragma unroll
  for (int t = 0; t < 2; ++t) {
    int c = n0 + 16 * t + r;
    const v8f& acc = t ? acc1 : acc0;
    if (c < N) {
      float bv = bias ? bias[c] : 0.0f;
#pragma unroll
      for (int v = 0; v < 8; ++v) {
        int row = m0 + v + 8 * g;
        if (row < M) {
          float o = acc[v] + bv;
          if (act_relu) o = fmaxf(o, 0.0f);
          if (C)  C[(size_t)row * N + c] = o;
          if (Ch) Ch[(size_t)row * N + c] = (_Float16)o;
        }
      }
    }
  }
}

// f32 [M,K] -> f16 [*, Kp], zero-padded. grid = (rows_padded, ceil(Kp/256))
__global__ void k_cvt_a(const float* __restrict__ A, _Float16* __restrict__ Ah,
                        int M, int K, int Kp) {
  int row = blockIdx.x;
  int k = blockIdx.y * 256 + threadIdx.x;
  if (k >= Kp) return;
  float v = (row < M && k < K) ? A[(size_t)row * K + k] : 0.0f;
  Ah[(size_t)row * Kp + k] = (_Float16)v;
}

// W[K,N] f32 -> Wt[Np,Kp] f16 (transpose + pad). grid = (Np, ceil(Kp/256))
__global__ void k_cvt_w(const float* __restrict__ W, _Float16* __restrict__ Wt,
                        int K, int N, int Kp) {
  int n = blockIdx.x;
  int k = blockIdx.y * 256 + threadIdx.x;
  if (k >= Kp) return;
  float v = (n < N && k < K) ? W[(size_t)k * N + n] : 0.0f;
  Wt[(size_t)n * Kp + k] = (_Float16)v;
}

// rel_s[l][r] = rel_emb[r] @ (lin_edge_w[l] @ att_edge[l])
__global__ __launch_bounds__(64) void k_rel(
    const float* __restrict__ rel_emb,
    const float* __restrict__ lew1, const float* __restrict__ ae1,
    const float* __restrict__ lew2, const float* __restrict__ ae2,
    float* __restrict__ rs1, float* __restrict__ rs2) {
  __shared__ float v1[32], v2[32];
  int t = threadIdx.x;
  if (t < 32) {
    float s = 0.0f;
    for (int j = 0; j < 128; ++j) s += lew1[t * 128 + j] * ae1[j];
    v1[t] = s;
  } else {
    int d = t - 32;
    float s = 0.0f;
    for (int j = 0; j < 128; ++j) s += lew2[d * 128 + j] * ae2[j];
    v2[d] = s;
  }
  __syncthreads();
  if (t < 26) {
    float s = 0.0f;
    for (int d = 0; d < 32; ++d) s += rel_emb[t * 32 + d] * v1[d];
    rs1[t] = s;
  } else if (t >= 32 && t < 58) {
    int rr = t - 32;
    float s = 0.0f;
    for (int d = 0; d < 32; ++d) s += rel_emb[rr * 32 + d] * v2[d];
    rs2[rr] = s;
  }
}

__global__ void k_zero(float* __restrict__ p, long n) {
  long i = (long)blockIdx.x * blockDim.x + threadIdx.x;
  if (i < n) p[i] = 0.0f;
}

__global__ void k_cnt_sum(const int* __restrict__ ei, const int* __restrict__ ea,
                          const float* __restrict__ rs1, const float* __restrict__ rs2,
                          float* __restrict__ cnt, float* __restrict__ s1,
                          float* __restrict__ s2, int E) {
  int e = blockIdx.x * blockDim.x + threadIdx.x;
  if (e >= E) return;
  int d = ei[E + e];
  int a = ea[e];
  atomicAdd(&cnt[d], 1.0f);
  atomicAdd(&s1[d], rs1[a]);
  atomicAdd(&s2[d], rs2[a]);
}

__global__ void k_node_pre(const float* __restrict__ Hlin,
                           const float* __restrict__ att_s, const float* __restrict__ att_d,
                           const float* __restrict__ sum_s, const float* __restrict__ cnt,
                           float* __restrict__ aS, float* __restrict__ aD,
                           float* __restrict__ aLoop, unsigned* __restrict__ amax, int n) {
  int i = blockIdx.x * blockDim.x + threadIdx.x;
  if (i >= n) return;
  const float4* hp = (const float4*)(Hlin + (size_t)i * NODE_DIM);
  const float4* sp = (const float4*)att_s;
  const float4* dp = (const float4*)att_d;
  float as = 0.0f, ad = 0.0f;
#pragma unroll 8
  for (int k = 0; k < NODE_DIM / 4; ++k) {
    float4 h = hp[k], s = sp[k], d = dp[k];
    as += h.x * s.x + h.y * s.y + h.z * s.z + h.w * s.w;
    ad += h.x * d.x + h.y * d.y + h.z * d.z + h.w * d.w;
  }
  float c = cnt[i];
  float loop_s = sum_s[i] / fmaxf(c, 1.0f);
  float al = leaky(as + ad + loop_s);
  aS[i] = as; aD[i] = ad; aLoop[i] = al;
  amax[i] = f2ord(al);
}

__global__ void k_edge_alpha(const int* __restrict__ ei, const int* __restrict__ ea,
                             const float* __restrict__ aS, const float* __restrict__ aD,
                             const float* __restrict__ rs, float* __restrict__ alpha,
                             unsigned* __restrict__ amax, int E) {
  int e = blockIdx.x * blockDim.x + threadIdx.x;
  if (e >= E) return;
  int s = ei[e], d = ei[E + e];
  float a = leaky(aS[s] + aD[d] + rs[ea[e]]);
  alpha[e] = a;
  atomicMax(&amax[d], f2ord(a));
}

__global__ void k_node_mid(const float* __restrict__ aLoop, const unsigned* __restrict__ amax,
                           float* __restrict__ denom, int n) {
  int i = blockIdx.x * blockDim.x + threadIdx.x;
  if (i >= n) return;
  denom[i] = __expf(aLoop[i] - ord2f(amax[i]));
}

__global__ void k_edge_exp(const int* __restrict__ ei, float* __restrict__ alpha,
                           const unsigned* __restrict__ amax, float* __restrict__ denom, int E) {
  int e = blockIdx.x * blockDim.x + threadIdx.x;
  if (e >= E) return;
  int d = ei[E + e];
  float ex = __expf(alpha[e] - ord2f(amax[d]));
  alpha[e] = ex;
  atomicAdd(&denom[d], ex);
}

__global__ void k_node_self(const float* __restrict__ Hlin, const float* __restrict__ aLoop,
                            const unsigned* __restrict__ amax, const float* __restrict__ denom,
                            float* __restrict__ Hout, int n) {
  int i = blockIdx.x * blockDim.x + threadIdx.x;
  if (i >= n) return;
  float w = __expf(aLoop[i] - ord2f(amax[i])) / denom[i];
  const float4* hp = (const float4*)(Hlin + (size_t)i * NODE_DIM);
  float4* op = (float4*)(Hout + (size_t)i * NODE_DIM);
#pragma unroll 8
  for (int k = 0; k < NODE_DIM / 4; ++k) {
    float4 h = hp[k];
    op[k] = make_float4(h.x * w, h.y * w, h.z * w, h.w * w);
  }
}

__global__ __launch_bounds__(256) void k_edge_agg(
    const int* __restrict__ ei, const float* __restrict__ alpha,
    const float* __restrict__ denom, const float* __restrict__ Hlin,
    float* __restrict__ Hout, int E) {
  int eid = blockIdx.x * 8 + (threadIdx.x >> 5);
  int lane = threadIdx.x & 31;
  if (eid >= E) return;
  int s = ei[eid], d = ei[E + eid];
  float w = alpha[eid] / denom[d];
  float4 h = *(const float4*)(Hlin + (size_t)s * NODE_DIM + lane * 4);
  float* o = Hout + (size_t)d * NODE_DIM + lane * 4;
  atomicAdd(o + 0, h.x * w);
  atomicAdd(o + 1, h.y * w);
  atomicAdd(o + 2, h.z * w);
  atomicAdd(o + 3, h.w * w);
}

// relu(Hout + bias) -> H (f32, for pool pass) and Hh (f16, for GEMM)
__global__ void k_node_fin(const float* __restrict__ Hout, const float* __restrict__ bias,
                           float* __restrict__ H, _Float16* __restrict__ Hh, long total) {
  long t = (long)blockIdx.x * blockDim.x + threadIdx.x;
  if (t >= total) return;
  float v = Hout[t] + bias[t & (NODE_DIM - 1)];
  v = v > 0.0f ? v : 0.0f;
  H[t] = v;
  Hh[t] = (_Float16)v;
}

// sorted-batch pooling: wave accumulates 32 consecutive nodes, flushes one
// atomic burst per graph transition
__global__ __launch_bounds__(256) void k_pool(const float* __restrict__ H,
                                              const int* __restrict__ batch,
                                              float* __restrict__ pooled,
                                              float* __restrict__ pcnt, int n) {
  int wid = blockIdx.x * 8 + (threadIdx.x >> 5);
  int lane = threadIdx.x & 31;
  int n0 = wid * 32;
  if (n0 >= n) return;
  int end = n0 + 32; if (end > n) end = n;
  float ax = 0, ay = 0, az = 0, aw = 0, c = 0;
  int cur = batch[n0];
  for (int i = n0; i < end; ++i) {
    int b = batch[i];
    if (b != cur) {
      float* p = pooled + (size_t)cur * NODE_DIM + lane * 4;
      atomicAdd(p + 0, ax); atomicAdd(p + 1, ay);
      atomicAdd(p + 2, az); atomicAdd(p + 3, aw);
      if (lane == 0) atomicAdd(&pcnt[cur], c);
      ax = ay = az = aw = c = 0.0f;
      cur = b;
    }
    float4 h = *(const float4*)(H + (size_t)i * NODE_DIM + lane * 4);
    ax += h.x; ay += h.y; az += h.z; aw += h.w;
    c += 1.0f;
  }
  float* p = pooled + (size_t)cur * NODE_DIM + lane * 4;
  atomicAdd(p + 0, ax); atomicAdd(p + 1, ay);
  atomicAdd(p + 2, az); atomicAdd(p + 3, aw);
  if (lane == 0) atomicAdd(&pcnt[cur], c);
}

__global__ void k_pool_mean(float* __restrict__ pooled, const float* __restrict__ pcnt, int total) {
  int t = blockIdx.x * blockDim.x + threadIdx.x;
  if (t >= total) return;
  pooled[t] /= fmaxf(pcnt[t / NODE_DIM], 1.0f);
}

extern "C" void kernel_launch(void* const* d_in, const int* in_sizes, int n_in,
                              void* d_out, int out_size, void* d_ws, size_t ws_size,
                              hipStream_t stream) {
  const float* x      = (const float*)d_in[0];
  const int*   ei     = (const int*)d_in[1];
  const int*   ea     = (const int*)d_in[2];
  const int*   batch  = (const int*)d_in[3];
  const float* enc_w1 = (const float*)d_in[4];
  const float* enc_b1 = (const float*)d_in[5];
  const float* enc_w2 = (const float*)d_in[6];
  const float* enc_b2 = (const float*)d_in[7];
  const float* rel_emb = (const float*)d_in[8];
  const float* lin_w[2]  = {(const float*)d_in[9],  (const float*)d_in[15]};
  const float* lew[2]    = {(const float*)d_in[10], (const float*)d_in[16]};
  const float* att_s[2]  = {(const float*)d_in[11], (const float*)d_in[17]};
  const float* att_d[2]  = {(const float*)d_in[12], (const float*)d_in[18]};
  const float* att_e[2]  = {(const float*)d_in[13], (const float*)d_in[19]};
  const float* cbias[2]  = {(const float*)d_in[14], (const float*)d_in[20]};
  const float* ro_w1 = (const float*)d_in[21];
  const float* ro_b1 = (const float*)d_in[22];
  const float* ro_w2 = (const float*)d_in[23];
  const float* ro_b2 = (const float*)d_in[24];

  const int N = in_sizes[3];          // N_NODES (multiple of 16 here)
  const int E = in_sizes[2];          // N_EDGES
  const int OBJ = in_sizes[0] / N;    // 518
  const int OBJp = (OBJ + 31) & ~31;  // 544

  float* ws = (float*)d_ws;
  size_t o = 0;
  auto take = [&](size_t nfloats) {
    float* p = ws + o;
    o += (nfloats + 63) & ~(size_t)63;  // 256B-align every buffer
    return p;
  };
  // f32 buffers
  float* H     = take((size_t)N * NODE_DIM);
  float* Hlin  = take((size_t)N * NODE_DIM);
  float* Hout  = take((size_t)N * NODE_DIM);
  float* aS    = take(N);
  float* aD    = take(N);
  float* aLoop = take(N);
  unsigned* amax = (unsigned*)take(N);
  float* denom = take(N);
  float* cnt   = take(3 * (size_t)N);  // cnt, sum1, sum2 contiguous
  float* sum1  = cnt + N;
  float* sum2  = cnt + 2 * (size_t)N;
  const float* sum_l[2] = {sum1, sum2};
  float* alpha = take(E);
  float* rs1   = take(32);
  float* rs2   = take(32);
  const float* rs_l[2] = {rs1, rs2};
  float* pooled = take(64 * NODE_DIM + 64);  // pooled + pcnt contiguous
  float* pcnt   = pooled + 64 * NODE_DIM;
  // f16 buffers (sizes in halfs -> /2 floats); Wt buffers padded to Np=128k
  _Float16* Xh   = (_Float16*)take((size_t)N * OBJp / 2);
  _Float16* H1h  = (_Float16*)take((size_t)N * 256 / 2);
  _Float16* Hh   = (_Float16*)take((size_t)N * NODE_DIM / 2);
  _Float16* w1t  = (_Float16*)take(256 * (size_t)OBJp / 2);  // [256, OBJp]
  _Float16* w2t  = (_Float16*)take(128 * 256 / 2);           // [128, 256]
  _Float16* lwt[2];
  lwt[0] = (_Float16*)take(128 * 128 / 2);
  lwt[1] = (_Float16*)take(128 * 128 / 2);
  _Float16* rw1t = (_Float16*)take(128 * 128 / 2);           // [128, 128] (N=64 padded)
  _Float16* rw2t = (_Float16*)take(128 * 64 / 2);            // [128, 64]  (N=32 padded)
  _Float16* pooledh = (_Float16*)take(64 * NODE_DIM / 2);
  _Float16* rohh    = (_Float16*)take(64 * 64 / 2);

  auto nb = [](long n_, int b_) { return (unsigned)((n_ + b_ - 1) / b_); };

  // ---- precompute: f16 conversions, relation scalars, edge stats ----
  k_cvt_a<<<dim3(N, nb(OBJp, 256)), 256, 0, stream>>>(x, Xh, N, OBJ, OBJp);
  k_cvt_w<<<dim3(256, nb(OBJp, 256)), 256, 0, stream>>>(enc_w1, w1t, OBJ, 256, OBJp);
  k_cvt_w<<<dim3(128, 1), 256, 0, stream>>>(enc_w2, w2t, 256, 128, 256);
  k_cvt_w<<<dim3(128, 1), 256, 0, stream>>>(lin_w[0], lwt[0], 128, 128, 128);
  k_cvt_w<<<dim3(128, 1), 256, 0, stream>>>(lin_w[1], lwt[1], 128, 128, 128);
  k_cvt_w<<<dim3(128, 1), 256, 0, stream>>>(ro_w1, rw1t, 128, 64, 128);
  k_cvt_w<<<dim3(128, 1), 256, 0, stream>>>(ro_w2, rw2t, 64, 32, 64);

  k_rel<<<1, 64, 0, stream>>>(rel_emb, lew[0], att_e[0], lew[1], att_e[1], rs1, rs2);
  k_zero<<<nb(3L * N, 256), 256, 0, stream>>>(cnt, 3L * N);
  k_cnt_sum<<<nb(E, 256), 256, 0, stream>>>(ei, ea, rs1, rs2, cnt, sum1, sum2, E);

  // ---- node encoder (WMMA) ----
  k_gemm_wmma<<<dim3(nb(N, 16), 2), 128, 0, stream>>>(
      Xh, w1t, enc_b1, nullptr, H1h, N, 256, OBJp, 1);
  k_gemm_wmma<<<dim3(nb(N, 16), 1), 128, 0, stream>>>(
      H1h, w2t, enc_b2, nullptr, Hh, N, 128, 256, 0);

  // ---- two GAT layers ----
  for (int l = 0; l < 2; ++l) {
    k_gemm_wmma<<<dim3(nb(N, 16), 1), 128, 0, stream>>>(
        Hh, lwt[l], nullptr, Hlin, nullptr, N, 128, 128, 0);
    k_node_pre<<<nb(N, 256), 256, 0, stream>>>(Hlin, att_s[l], att_d[l], sum_l[l], cnt,
                                               aS, aD, aLoop, amax, N);
    k_edge_alpha<<<nb(E, 256), 256, 0, stream>>>(ei, ea, aS, aD, rs_l[l], alpha, amax, E);
    k_node_mid<<<nb(N, 256), 256, 0, stream>>>(aLoop, amax, denom, N);
    k_edge_exp<<<nb(E, 256), 256, 0, stream>>>(ei, alpha, amax, denom, E);
    k_node_self<<<nb(N, 256), 256, 0, stream>>>(Hlin, aLoop, amax, denom, Hout, N);
    k_edge_agg<<<nb(E, 8), 256, 0, stream>>>(ei, alpha, denom, Hlin, Hout, E);
    k_node_fin<<<nb((long)N * NODE_DIM, 256), 256, 0, stream>>>(
        Hout, cbias[l], H, Hh, (long)N * NODE_DIM);
  }

  // ---- mean pool + readout MLP (WMMA) ----
  k_zero<<<nb(64 * NODE_DIM + 64, 256), 256, 0, stream>>>(pooled, 64 * NODE_DIM + 64);
  k_pool<<<nb(nb(N, 32), 8), 256, 0, stream>>>(H, batch, pooled, pcnt, N);
  k_pool_mean<<<nb(64 * NODE_DIM, 256), 256, 0, stream>>>(pooled, pcnt, 64 * NODE_DIM);
  k_cvt_a<<<dim3(64, 1), 256, 0, stream>>>(pooled, pooledh, 64, 128, 128);
  k_gemm_wmma<<<dim3(4, 1), 128, 0, stream>>>(
      pooledh, rw1t, ro_b1, nullptr, rohh, 64, 64, 128, 1);
  k_gemm_wmma<<<dim3(4, 1), 128, 0, stream>>>(
      rohh, rw2t, ro_b2, (float*)d_out, nullptr, 64, 32, 64, 0);
}
